// Intra_slice_Classaware_Interaction_20383914787107
// MI455X (gfx1250) — matrix operand order repaired and checked by reading, not compiled, not run on hardware
//
#include <hip/hip_runtime.h>

#define B_ 8
#define C_ 64
#define Hh 192
#define Ww 192
#define HW 36864
#define NC 2
#define NS 8
#define NMAP 16      // B_*NC
#define NCHUNK 96
#define ETILES 4     // 64/16
#define PTILES 2304  // HW/16
#define CTILES 4
#define KTOT 576     // 64*9
#define PADW 194
#define PADHW 37636  // 194*194

typedef __attribute__((ext_vector_type(16))) _Float16 v16h;
typedef __attribute__((ext_vector_type(8)))  _Float16 v8h;
typedef __attribute__((ext_vector_type(8)))  float    v8f;
typedef __attribute__((ext_vector_type(2)))  float    v2f;

__device__ __forceinline__ float sigmoidf(float x) { return 1.0f / (1.0f + __expf(-x)); }

// ---------------------------------------------------------------------------
// 0) copy x into zero-padded (194x194) per-map layout (borders pre-zeroed)
// ---------------------------------------------------------------------------
__global__ void xpad_kernel(const float* __restrict__ x, float* __restrict__ xp) {
  int idx = blockIdx.x * blockDim.x + threadIdx.x;   // B_*C_*HW
  int p = idx % HW;
  int be = idx / HW;
  int h = p / Ww, w = p % Ww;
  xp[(size_t)be * PADHW + (h + 1) * PADW + (w + 1)] = x[idx];
}

// ---------------------------------------------------------------------------
// 1) space[(b*NC+c)*NS + s][hw]: s=0 -> sigmoid(score); s>=1 -> sigmoid(conv3x3)
// ---------------------------------------------------------------------------
__global__ void space_kernel(const float* __restrict__ score,
                             const float* __restrict__ Wc,
                             float* __restrict__ space) {
  int m = blockIdx.y;                          // b*NC + c
  int p = blockIdx.x * blockDim.x + threadIdx.x;
  int h = p / Ww, w = p % Ww;
  const float* s0 = score + (size_t)m * HW;
  float sig[9];
#pragma unroll
  for (int k = 0; k < 9; ++k) {
    int dy = k / 3 - 1, dx = k % 3 - 1;
    int hh = h + dy, ww2 = w + dx;
    sig[k] = ((unsigned)hh < Hh && (unsigned)ww2 < Ww) ? sigmoidf(s0[hh * Ww + ww2]) : 0.0f;
  }
  float* sp = space + (size_t)m * NS * HW + p;
  sp[0] = sig[4];
#pragma unroll
  for (int s = 1; s < NS; ++s) {
    float acc = 0.f;
#pragma unroll
    for (int k = 0; k < 9; ++k) acc += Wc[(s - 1) * 9 + k] * sig[k];
    sp[(size_t)s * HW] = sigmoidf(acc);
  }
}

// ---------------------------------------------------------------------------
// 2) per-map spatial sums (deterministic tree reduce)
// ---------------------------------------------------------------------------
__global__ void space_sum_kernel(const float* __restrict__ space, float* __restrict__ sums) {
  __shared__ float sm[256];
  int m = blockIdx.x, tid = threadIdx.x;
  const float* sp = space + (size_t)m * HW;
  float s = 0.f;
  for (int i = tid; i < HW; i += 256) s += sp[i];
  sm[tid] = s; __syncthreads();
  for (int o = 128; o > 0; o >>= 1) { if (tid < o) sm[tid] += sm[tid + o]; __syncthreads(); }
  if (tid == 0) sums[m] = sm[0];
}

// ---------------------------------------------------------------------------
// 3) corr partials via V_WMMA_F32_16X16X4_F32 (exact f32 GEMM)
//    M=16 (s,c)-maps, N=16 e-channels, K over pixels; 9 shift-accumulators.
//    Padded x -> all loads unconditional, immediate offsets, no divisions.
// ---------------------------------------------------------------------------
__global__ void corr_part_kernel(const float* __restrict__ xpad,
                                 const float* __restrict__ space,
                                 float* __restrict__ part) {
  int wid = threadIdx.x >> 5;
  int lane = threadIdx.x & 31;
  int unit = blockIdx.x * 8 + wid;             // 0..3071
  int chunk = unit % NCHUNK;
  int et = (unit / NCHUNK) % ETILES;
  int b  = unit / (NCHUNK * ETILES);
  int half = lane >> 4, n = lane & 15;
  int cM = n >> 3, sM = n & 7;                 // M = c*8 + s
  const float* spA = space + (size_t)((b * NC + cM) * NS + sM) * HW;
  const float* xB  = xpad + (size_t)(b * C_ + et * 16 + n) * PADHW;

  v8f acc[9] = {};
#pragma unroll
  for (int r2 = 0; r2 < 2; ++r2) {
    int h = chunk * 2 + r2;
    const float* spRow = spA + h * Ww + half * 2;            // A lane K = 2*half+{0,1}
    const float* xRow  = xB + (h + 1) * PADW + half * 2 + 1; // padded center
    for (int s48 = 0; s48 < 48; ++s48) {
      v2f a = *(const v2f*)(spRow + s48 * 4);                // 8B aligned
      const float* xc = xRow + s48 * 4;
#pragma unroll
      for (int sh = 0; sh < 9; ++sh) {
        int off = (sh / 3 - 1) * PADW + (sh % 3 - 1);        // constant imm offset
        v2f bb; bb[0] = xc[off]; bb[1] = xc[off + 1];
        acc[sh] = __builtin_amdgcn_wmma_f32_16x16x4_f32(false, a, false, bb,
                                                        (short)0, acc[sh], false, false);
      }
    }
  }
  size_t base = ((size_t)(b * ETILES + et) * NCHUNK + chunk) * 9 * 256;
#pragma unroll
  for (int sh = 0; sh < 9; ++sh)
#pragma unroll
    for (int r = 0; r < 8; ++r) {
      int M = r + half * 8;
      part[base + (size_t)sh * 256 + M * 16 + n] = acc[sh][r];
    }
}

// ---------------------------------------------------------------------------
// 4) fixed-order chunk reduction -> corr[((b*16+M)*64+e)*9+sh]
// ---------------------------------------------------------------------------
__global__ void corr_reduce_kernel(const float* __restrict__ part, float* __restrict__ corr) {
  int idx = blockIdx.x * blockDim.x + threadIdx.x;   // 73728
  int sh = idx % 9;
  int e  = (idx / 9) % 64;
  int M  = (idx / (9 * 64)) % 16;
  int b  = idx / (9 * 64 * 16);
  int et = e >> 4, el = e & 15;
  const size_t stride = 9 * 256;
  const float* p = part + (size_t)(b * ETILES + et) * NCHUNK * stride
                        + (size_t)sh * 256 + M * 16 + el;
  float s = 0.f;
  for (int ch = 0; ch < NCHUNK; ++ch) s += p[(size_t)ch * stride];
  corr[idx] = s;
}

// ---------------------------------------------------------------------------
// 5) tok2[b,c,d] = sum_s Wcs[s]/sum_sp * sum_{e,k} W_xspace[s*64+d,e,k]*corr
// ---------------------------------------------------------------------------
__global__ void tok_kernel(const float* __restrict__ corr, const float* __restrict__ Wx,
                           const float* __restrict__ Wcs, const float* __restrict__ sums,
                           float* __restrict__ tok2) {
  int bc = blockIdx.x;                // b*NC + c
  int b = bc / NC, c = bc % NC;
  int d = threadIdx.x;                // 0..63
  float out = 0.f;
  for (int s = 0; s < NS; ++s) {
    int M = c * NS + s;
    const float* cr = corr + (size_t)(b * 16 + M) * KTOT;
    const float* wx = Wx + (size_t)(s * 64 + d) * KTOT;
    float acc = 0.f;
    for (int i = 0; i < KTOT; ++i) acc += wx[i] * cr[i];
    out += Wcs[s] * acc / sums[bc * NS + s];
  }
  tok2[bc * 64 + d] = out;
}

// ---------------------------------------------------------------------------
// 6) info16p[b][padded hw][d] (f16, channel-contiguous, zero borders)
//    = g0*tok2[b,0,d] + g1*tok2[b,1,d]
// ---------------------------------------------------------------------------
__global__ void info_kernel(const float* __restrict__ score,
                            const float* __restrict__ tok2,
                            const float* __restrict__ Wcomb,
                            const float* __restrict__ Wcval,
                            const float* __restrict__ Wattn,
                            _Float16* __restrict__ info16) {
  __shared__ float tk[128];
  __shared__ float wg[16];
  int tid = threadIdx.x;
  int b = blockIdx.x / (HW / 256);
  int p = (blockIdx.x % (HW / 256)) * 256 + tid;
  if (tid < 128) tk[tid] = tok2[b * 128 + tid];
  if (tid < 8) { wg[tid] = Wcomb[tid] * Wcval[tid]; wg[8 + tid] = Wattn[tid]; }
  __syncthreads();
  float g[2];
#pragma unroll
  for (int c = 0; c < 2; ++c) {
    float pc = sigmoidf(score[(size_t)(b * NC + c) * HW + p]);
    float s = 0.f;
#pragma unroll
    for (int j = 0; j < 8; ++j) s += wg[j] * sigmoidf(pc * wg[8 + j]);
    g[c] = s;
  }
  int h = p / Ww, w = p % Ww;
  _Float16* out = info16 + ((size_t)b * PADHW + (h + 1) * PADW + (w + 1)) * 64;
#pragma unroll
  for (int d = 0; d < 64; ++d)
    out[d] = (_Float16)(g[0] * tk[d] + g[1] * tk[64 + d]);
}

// ---------------------------------------------------------------------------
// 7) reorder W_single (dO,e,k) f32 -> Wl[dO][k*64+e] f16
// ---------------------------------------------------------------------------
__global__ void wprep_kernel(const float* __restrict__ Ws, _Float16* __restrict__ Wl) {
  int idx = blockIdx.x * blockDim.x + threadIdx.x;   // 36864
  int K = idx % KTOT, dO = idx / KTOT;
  int e = K & 63, k = K >> 6;
  Wl[idx] = (_Float16)Ws[(size_t)(dO * 64 + e) * 9 + k];
}

// ---------------------------------------------------------------------------
// 8) 3x3 conv 64->64, implicit GEMM via V_WMMA_F32_16X16X32_F16
//    per wave: 16 out-channels x 16 row-adjacent pixels, K = 576 in 18 chunks
//    Padded info -> B fragment = one unconditional 32B contiguous load.
// ---------------------------------------------------------------------------
__global__ void conv_kernel(const _Float16* __restrict__ info16,
                            const _Float16* __restrict__ Wl,
                            float* __restrict__ y) {
  int wid = threadIdx.x >> 5;
  int lane = threadIdx.x & 31;
  int unit = blockIdx.x * 8 + wid;             // 0..73727
  int pt = unit % PTILES;
  int ct = (unit / PTILES) % CTILES;
  int b  = unit / (PTILES * CTILES);
  int h = pt / (Ww / 16);
  int wseg = (pt % (Ww / 16)) * 16;
  int half = lane >> 4, n = lane & 15;
  int w = wseg + n;
  const _Float16* arow = Wl + (size_t)(ct * 16 + n) * KTOT;
  const _Float16* brow = info16 + ((size_t)b * PADHW + (h + 1) * PADW + (w + 1)) * 64;
  v8f c = {};
#pragma unroll
  for (int t = 0; t < 18; ++t) {
    int k = t >> 1;
    int dy = k / 3 - 1, dx = k % 3 - 1;
    // A: contiguous k order (code's choice) -> lanes<16: K0..15, lanes>=16: K16..31 (matches B)
    int koffA = t * 32 + half * 16;
    union { v16h v; v8h h2[2]; } ua;
    ua.v = *(const v16h*)(arow + koffA);
    // B: contiguous k order, same as A -> lanes<16: K0..15, lanes>=16: K16..31 (contiguous e range)
    int eb = ((t & 1) << 5) + (half << 4);
    v16h bfrag = *(const v16h*)(brow + ((size_t)(dy * PADW + dx)) * 64 + eb);
    c = __builtin_amdgcn_wmma_f32_16x16x32_f16(false, ua.v, false, bfrag,
                                               (short)0, c, false, false);
  }
#pragma unroll
  for (int r = 0; r < 8; ++r) {
    int ch = ct * 16 + r + half * 8;           // C/D layout: lanes>=16 -> M=r+8
    y[(size_t)(b * C_ + ch) * HW + h * Ww + wseg + n] = c[r];
  }
}

// ---------------------------------------------------------------------------
// 9) per-channel sum / sumsq over (b,H,W) (deterministic)
// ---------------------------------------------------------------------------
__global__ void bnstat_kernel(const float* __restrict__ y, float* __restrict__ stats) {
  __shared__ float sm[256];
  int ch = blockIdx.x, tid = threadIdx.x;
  float s = 0.f, s2 = 0.f;
  for (int i = tid; i < B_ * HW; i += 256) {
    int b = i / HW, p = i % HW;
    float v = y[(size_t)(b * C_ + ch) * HW + p];
    s += v; s2 += v * v;
  }
  sm[tid] = s; __syncthreads();
  for (int o = 128; o > 0; o >>= 1) { if (tid < o) sm[tid] += sm[tid + o]; __syncthreads(); }
  if (tid == 0) stats[ch] = sm[0];
  __syncthreads();
  sm[tid] = s2; __syncthreads();
  for (int o = 128; o > 0; o >>= 1) { if (tid < o) sm[tid] += sm[tid + o]; __syncthreads(); }
  if (tid == 0) stats[64 + ch] = sm[0];
}

// ---------------------------------------------------------------------------
// 10) out = x + relu(BN(y)) , in place on d_out
// ---------------------------------------------------------------------------
__global__ void final_kernel(const float* __restrict__ x,
                             const float* __restrict__ stats,
                             const float* __restrict__ gamma,
                             const float* __restrict__ beta,
                             float* __restrict__ out) {
  size_t idx = (size_t)blockIdx.x * blockDim.x + threadIdx.x;
  int ch = (int)((idx / HW) % C_);
  const float Ninv = 1.0f / (float)(B_ * HW);
  float mean = stats[ch] * Ninv;
  float var  = stats[64 + ch] * Ninv - mean * mean;
  float v = out[idx];
  float r = (v - mean) * rsqrtf(var + 1e-5f) * gamma[ch] + beta[ch];
  out[idx] = x[idx] + fmaxf(r, 0.f);
}

extern "C" void kernel_launch(void* const* d_in, const int* in_sizes, int n_in,
                              void* d_out, int out_size, void* d_ws, size_t ws_size,
                              hipStream_t stream) {
  (void)in_sizes; (void)n_in; (void)out_size; (void)ws_size;
  const float* x     = (const float*)d_in[0];
  const float* score = (const float*)d_in[1];
  const float* Wcsp  = (const float*)d_in[2];
  const float* Wx    = (const float*)d_in[3];
  const float* Wcs   = (const float*)d_in[4];
  const float* Wcval = (const float*)d_in[5];
  const float* Wattn = (const float*)d_in[6];
  const float* Wcomb = (const float*)d_in[7];
  const float* Wsing = (const float*)d_in[8];
  const float* gamma = (const float*)d_in[9];
  const float* beta  = (const float*)d_in[10];
  float* y = (float*)d_out;

  float* ws    = (float*)d_ws;
  float* space = ws;                                              // 4,718,592 f
  float* sums  = space + (size_t)NMAP * NS * HW;                  // 128 f
  float* part  = sums + 128;                                      // 7,077,888 f
  float* corr  = part + (size_t)B_ * ETILES * NCHUNK * 9 * 256;   // 73,728 f
  float* tok2  = corr + 73728;                                    // 1,024 f
  float* stats = tok2 + 1024;                                     // 128 f
  _Float16* Wl = (_Float16*)(stats + 128);                        // 36,864 h
  // union region: xpad (f32, dead after corr_part) / info16p (f16, used after)
  void* uni = (void*)(Wl + 36864);
  float* xpad = (float*)uni;                   // 19,269,632 f = 77.1 MB
  _Float16* info16 = (_Float16*)uni;           // 19,269,632 h = 38.5 MB

  hipMemsetAsync(xpad, 0, (size_t)B_ * C_ * PADHW * sizeof(float), stream);
  xpad_kernel<<<(B_ * C_ * HW) / 256, 256, 0, stream>>>(x, xpad);
  space_kernel<<<dim3(HW / 256, NMAP), 256, 0, stream>>>(score, Wcsp, space);
  space_sum_kernel<<<NMAP * NS, 256, 0, stream>>>(space, sums);
  corr_part_kernel<<<(B_ * ETILES * NCHUNK) / 8, 256, 0, stream>>>(xpad, space, part);
  corr_reduce_kernel<<<73728 / 256, 256, 0, stream>>>(part, corr);
  tok_kernel<<<B_ * NC, 64, 0, stream>>>(corr, Wx, Wcs, sums, tok2);
  hipMemsetAsync(info16, 0, (size_t)B_ * PADHW * C_ * sizeof(_Float16), stream);
  info_kernel<<<B_ * (HW / 256), 256, 0, stream>>>(score, tok2, Wcomb, Wcval, Wattn, info16);
  wprep_kernel<<<KTOT * C_ / 256, 256, 0, stream>>>(Wsing, Wl);
  conv_kernel<<<(B_ * CTILES * PTILES) / 8, 256, 0, stream>>>(info16, Wl, y);
  bnstat_kernel<<<C_, 256, 0, stream>>>(y, stats);
  final_kernel<<<(B_ * C_ * HW) / 256, 256, 0, stream>>>(x, stats, gamma, beta, y);
}
